// SimpleRNN_48335561949443
// MI455X (gfx1250) — matrix operand order repaired and checked by reading, not compiled, not run on hardware
//
#include <hip/hip_runtime.h>
#include <hip/hip_bf16.h>

// Problem sizes (fixed by the reference)
#define B_SZ   64
#define T_SZ   512
#define IN_SZ  512
#define H_SZ   1024
#define NWG    32            // one WG per 32-column N-strip
#define THR    128           // 4 waves per WG (one per M-tile)

// padded LDS row strides (stride mod 64 banks == 4 dwords -> balanced b128 reads)
#define LIN    (IN_SZ + 8)
#define LH     (H_SZ + 8)
#define LDS_ELEMS (32 * LIN + 3 * 32 * LH)
#define LDS_BYTES (LDS_ELEMS * 2)

typedef __bf16 bf16_t;
typedef __attribute__((ext_vector_type(16))) __bf16 v16bf;
typedef __attribute__((ext_vector_type(8)))  __bf16 v8bf;
typedef __attribute__((ext_vector_type(8)))  float  v8f;

// ---------------------------------------------------------------------------
// fp32 -> bf16 weight conversion (weights are [N][K] row-major, which is
// exactly the per-lane-contiguous k layout this code's B loads use)
__global__ void cvt_f32_bf16_kernel(const float* __restrict__ src,
                                    bf16_t* __restrict__ dst, int n) {
    int i = blockIdx.x * blockDim.x + threadIdx.x;
    if (i < n) dst[i] = (bf16_t)src[i];
}

// combined bias b = b_ih + b_hh for both layers
__global__ void bias_combine_kernel(const float* __restrict__ bi0, const float* __restrict__ bh0,
                                    const float* __restrict__ bi1, const float* __restrict__ bh1,
                                    float* __restrict__ b0, float* __restrict__ b1) {
    int i = blockIdx.x * blockDim.x + threadIdx.x;
    if (i < H_SZ) { b0[i] = bi0[i] + bh0[i]; b1[i] = bi1[i] + bh1[i]; }
}

// zero hidden-state ping-pong buffers (bf16: zero bits == +0) + barrier state
__global__ void init_state_kernel(unsigned int* __restrict__ p, int n, int* __restrict__ bar) {
    int i = blockIdx.x * blockDim.x + threadIdx.x;
    if (i < n) p[i] = 0u;
    if (i < 2) bar[i] = 0;
}

// ---------------------------------------------------------------------------
// hardware tanh if the gfx1250 trans-op builtin is exposed
__device__ __forceinline__ float fast_tanh(float v) {
#if __has_builtin(__builtin_amdgcn_tanhf)
    return __builtin_amdgcn_tanhf(v);
#else
    return tanhf(v);
#endif
}

// ---------------------------------------------------------------------------
// device-wide split barrier (count + generation), NWG workgroups
__device__ __forceinline__ void grid_sync(int* bar) {
    __syncthreads();
    if (threadIdx.x == 0) {
        __threadfence();
        int gen = __hip_atomic_load(&bar[1], __ATOMIC_RELAXED, __HIP_MEMORY_SCOPE_AGENT);
        int arrived = __hip_atomic_fetch_add(&bar[0], 1, __ATOMIC_ACQ_REL, __HIP_MEMORY_SCOPE_AGENT);
        if (arrived == NWG - 1) {
            __hip_atomic_store(&bar[0], 0, __ATOMIC_RELAXED, __HIP_MEMORY_SCOPE_AGENT);
            __hip_atomic_fetch_add(&bar[1], 1, __ATOMIC_RELEASE, __HIP_MEMORY_SCOPE_AGENT);
        } else {
            while (__hip_atomic_load(&bar[1], __ATOMIC_ACQUIRE, __HIP_MEMORY_SCOPE_AGENT) == gen)
                __builtin_amdgcn_s_sleep(2);
        }
        __threadfence();
    }
    __syncthreads();
}

// ---------------------------------------------------------------------------
// Copy this WG's 32 weight rows (columns blk*32..+31) into LDS using the
// CDNA5 async global->LDS path (no VGPR round-trip, tracked on ASYNCcnt).
// Rows padded to K+8 elements so strip loads are bank-balanced.
// All loop trip counts are exact multiples of THR -> EXEC is all-ones.
__device__ __forceinline__ void stage_weights_async(const bf16_t* __restrict__ gsrc,
                                                    bf16_t* ldst, int K, int blk) {
    const int chunks = K >> 3;                  // 16B chunks per row
    for (int idx = threadIdx.x; idx < 32 * chunks; idx += THR) {
        int c = idx / chunks;
        int q = idx - c * chunks;
        unsigned long long gaddr =
            (unsigned long long)(const void*)(gsrc + ((size_t)(blk * 32 + c)) * K + (q << 3));
        unsigned lds_off =
            (unsigned)(unsigned long long)(void*)(ldst + (size_t)c * (K + 8) + (q << 3));
        asm volatile("global_load_async_to_lds_b128 %0, %1, off"
                     :: "v"(lds_off), "v"(gaddr)
                     : "memory");
    }
}

// ---------------------------------------------------------------------------
// One K-span of a 16(M) x 32(N) strip.  A: bf16 activations from global
// (row mr).  B0/B1: weight rows resident in LDS (stride K+8).
// A fragment (16x32): elems 0..15 = K[kb+hf*16 .. +15]
// B fragment (32x16): elems 0..15 = K[kb+hf*16 .. +15]
// #pragma unroll 2 keeps kb a runtime induction variable so the compiler
// cannot hoist the whole invariant fragment stream and spill it to scratch.
__device__ __forceinline__ void gemm_span2(v8f& acc0, v8f& acc1,
                                           const bf16_t* __restrict__ aRow,
                                           const bf16_t* bRow0,
                                           const bf16_t* bRow1,
                                           int K, int hf) {
    #pragma unroll 2
    for (int kb = 0; kb < K; kb += 32) {
        v8bf alo = *(const v8bf*)(aRow + kb + hf * 16);
        v8bf ahi = *(const v8bf*)(aRow + kb + hf * 16 + 8);
        v16bf A = __builtin_shufflevector(alo, ahi, 0,1,2,3,4,5,6,7,8,9,10,11,12,13,14,15);

        const int kbb = kb + hf * 16;
        v8bf b0lo = *(const v8bf*)(bRow0 + kbb);
        v8bf b0hi = *(const v8bf*)(bRow0 + kbb + 8);
        v16bf B0 = __builtin_shufflevector(b0lo, b0hi, 0,1,2,3,4,5,6,7,8,9,10,11,12,13,14,15);
        v8bf b1lo = *(const v8bf*)(bRow1 + kbb);
        v8bf b1hi = *(const v8bf*)(bRow1 + kbb + 8);
        v16bf B1 = __builtin_shufflevector(b1lo, b1hi, 0,1,2,3,4,5,6,7,8,9,10,11,12,13,14,15);

        acc0 = __builtin_amdgcn_wmma_f32_16x16x32_bf16(false, A, false, B0, (short)0, acc0, false, false);
        acc1 = __builtin_amdgcn_wmma_f32_16x16x32_bf16(false, A, false, B1, (short)0, acc1, false, false);
    }
}

// cooperative conversion of one timestep of x into the bf16 ping-pong slot
__device__ __forceinline__ void cvt_x_step(const float* __restrict__ x,
                                           bf16_t* __restrict__ xdst, int t, int tid) {
    // NWG*THR = 4096 threads, 8 contiguous elements each -> 64*512 covered
    int base = tid << 3;                 // 0..32760
    int b = base >> 9;                   // / IN_SZ
    int i = base & (IN_SZ - 1);
    const float* src = x + ((size_t)b * T_SZ + t) * IN_SZ + i;
    float4 f0 = *(const float4*)(src);
    float4 f1 = *(const float4*)(src + 4);
    v8bf o;
    o[0] = (bf16_t)f0.x; o[1] = (bf16_t)f0.y; o[2] = (bf16_t)f0.z; o[3] = (bf16_t)f0.w;
    o[4] = (bf16_t)f1.x; o[5] = (bf16_t)f1.y; o[6] = (bf16_t)f1.z; o[7] = (bf16_t)f1.w;
    *(v8bf*)(xdst + base) = o;
}

// ---------------------------------------------------------------------------
// Persistent scan kernel: weights LDS-resident, all 512 timesteps inside,
// grid barriers between phases.
__global__ __launch_bounds__(THR, 1)
void rnn_scan_kernel(const float* __restrict__ x,
                     const bf16_t* __restrict__ Wc0, const bf16_t* __restrict__ Wh0,
                     const bf16_t* __restrict__ Wc1, const bf16_t* __restrict__ Wh1,
                     const float* __restrict__ b0, const float* __restrict__ b1,
                     bf16_t* __restrict__ xbuf,               // 2 x [B][IN] bf16
                     bf16_t* __restrict__ h0buf,              // 2 x [B][H] bf16
                     bf16_t* __restrict__ h1buf,              // 2 x [B][H] bf16
                     float* __restrict__ out, float* __restrict__ hfin,
                     int* __restrict__ bar) {
    extern __shared__ __align__(16) bf16_t ldsw[];
    bf16_t* Lc0 = ldsw;                       // 32 x LIN
    bf16_t* Lh0 = Lc0 + 32 * LIN;             // 32 x LH
    bf16_t* Lc1 = Lh0 + 32 * LH;              // 32 x LH
    bf16_t* Lh1 = Lc1 + 32 * LH;              // 32 x LH

    const int blk  = blockIdx.x;                        // N-strip: cols [32*blk, +32)
    const int lane = threadIdx.x & 31;
    const int hf   = lane >> 4;                         // wave half
    const int l16  = lane & 15;
    const int tid  = blk * THR + threadIdx.x;           // 0..4095
    const int m0   = (threadIdx.x >> 5) << 4;           // wave -> M-tile (0..3)
    const int mr   = m0 + l16;                          // A row handled by this lane
    const int nn0  = blk * 32 + l16;                    // strip tile 0 column
    const int nn1  = blk * 32 + 16 + l16;               // strip tile 1 column

    // ---- one-time: stage this strip's weights into LDS (async path) ----
    stage_weights_async(Wc0, Lc0, IN_SZ, blk);
    stage_weights_async(Wh0, Lh0, H_SZ, blk);
    stage_weights_async(Wc1, Lc1, H_SZ, blk);
    stage_weights_async(Wh1, Lh1, H_SZ, blk);

    const float bias00 = b0[nn0], bias01 = b0[nn1];
    const float bias10 = b1[nn0], bias11 = b1[nn1];
    const bf16_t* c0r0 = Lc0 + (size_t)l16 * LIN;
    const bf16_t* c0r1 = Lc0 + (size_t)(l16 + 16) * LIN;
    const bf16_t* h0r0 = Lh0 + (size_t)l16 * LH;
    const bf16_t* h0r1 = Lh0 + (size_t)(l16 + 16) * LH;
    const bf16_t* c1r0 = Lc1 + (size_t)l16 * LH;
    const bf16_t* c1r1 = Lc1 + (size_t)(l16 + 16) * LH;
    const bf16_t* h1r0 = Lh1 + (size_t)l16 * LH;
    const bf16_t* h1r1 = Lh1 + (size_t)(l16 + 16) * LH;

    // stage x_0 into xbuf[0]
    cvt_x_step(x, xbuf, 0, tid);
    // wait for async weight staging to land in LDS, then global barrier
    asm volatile("s_wait_asynccnt 0x0" ::: "memory");
    grid_sync(bar);

    for (int t = 0; t < T_SZ; ++t) {
        const int r = t & 1, w = r ^ 1;
        const bf16_t* xcur = xbuf + (size_t)r * (B_SZ * IN_SZ);
        const bf16_t* h0p  = h0buf + (size_t)r * (B_SZ * H_SZ);
        bf16_t*       h0c  = h0buf + (size_t)w * (B_SZ * H_SZ);
        const bf16_t* h1p  = h1buf + (size_t)r * (B_SZ * H_SZ);
        bf16_t*       h1c  = h1buf + (size_t)w * (B_SZ * H_SZ);

        // ---- layer 0: h0 = tanh(x_t @ W_ih0^T + h0_prev @ W_hh0^T + b0) ----
        v8f acc0 = {0.f,0.f,0.f,0.f,0.f,0.f,0.f,0.f};
        v8f acc1 = {0.f,0.f,0.f,0.f,0.f,0.f,0.f,0.f};
        gemm_span2(acc0, acc1, xcur + (size_t)mr * IN_SZ, c0r0, c0r1, IN_SZ, hf);
        gemm_span2(acc0, acc1, h0p  + (size_t)mr * H_SZ,  h0r0, h0r1, H_SZ,  hf);
        #pragma unroll
        for (int v = 0; v < 8; ++v) {
            int m = m0 + v + hf * 8;
            float v0 = fast_tanh(acc0[v] + bias00);
            float v1 = fast_tanh(acc1[v] + bias01);
            h0c[(size_t)m * H_SZ + nn0] = (bf16_t)v0;
            h0c[(size_t)m * H_SZ + nn1] = (bf16_t)v1;
            if (t == T_SZ - 1) {
                hfin[(size_t)m * H_SZ + nn0] = v0;
                hfin[(size_t)m * H_SZ + nn1] = v1;
            }
        }
        grid_sync(bar);

        // ---- layer 1: h1 = tanh(h0 @ W_ih1^T + h1_prev @ W_hh1^T + b1) ----
        acc0 = (v8f){0.f,0.f,0.f,0.f,0.f,0.f,0.f,0.f};
        acc1 = (v8f){0.f,0.f,0.f,0.f,0.f,0.f,0.f,0.f};
        gemm_span2(acc0, acc1, h0c + (size_t)mr * H_SZ, c1r0, c1r1, H_SZ, hf);
        gemm_span2(acc0, acc1, h1p + (size_t)mr * H_SZ, h1r0, h1r1, H_SZ, hf);
        #pragma unroll
        for (int v = 0; v < 8; ++v) {
            int m = m0 + v + hf * 8;
            float v0 = fast_tanh(acc0[v] + bias10);
            float v1 = fast_tanh(acc1[v] + bias11);
            h1c[(size_t)m * H_SZ + nn0] = (bf16_t)v0;
            h1c[(size_t)m * H_SZ + nn1] = (bf16_t)v1;
            out[((size_t)m * T_SZ + t) * H_SZ + nn0] = v0;
            out[((size_t)m * T_SZ + t) * H_SZ + nn1] = v1;
            if (t == T_SZ - 1) {
                hfin[(size_t)(B_SZ * H_SZ) + (size_t)m * H_SZ + nn0] = v0;
                hfin[(size_t)(B_SZ * H_SZ) + (size_t)m * H_SZ + nn1] = v1;
            }
        }
        // stage x_{t+1} while layer-1 results are produced; the barrier below
        // orders it before layer 0 of step t+1 consumes it.
        if (t + 1 < T_SZ)
            cvt_x_step(x, xbuf + (size_t)w * (B_SZ * IN_SZ), t + 1, tid);
        grid_sync(bar);
    }
}

// ---------------------------------------------------------------------------
extern "C" void kernel_launch(void* const* d_in, const int* in_sizes, int n_in,
                              void* d_out, int out_size, void* d_ws, size_t ws_size,
                              hipStream_t stream) {
    const float* x     = (const float*)d_in[0];
    const float* W_ih0 = (const float*)d_in[1];
    const float* W_hh0 = (const float*)d_in[2];
    const float* b_ih0 = (const float*)d_in[3];
    const float* b_hh0 = (const float*)d_in[4];
    const float* W_ih1 = (const float*)d_in[5];
    const float* W_hh1 = (const float*)d_in[6];
    const float* b_ih1 = (const float*)d_in[7];
    const float* b_hh1 = (const float*)d_in[8];
    // d_in[9]/d_in[10] (fc_w, fc_b) unused: reference discards fc output.

    float* out  = (float*)d_out;
    float* hfin = out + (size_t)B_SZ * T_SZ * H_SZ;   // h_final [2,B,H] after outputs

    // workspace layout
    char* ws = (char*)d_ws;
    size_t off = 0;
    bf16_t* Wc0 = (bf16_t*)(ws + off); off += (size_t)H_SZ * IN_SZ * sizeof(bf16_t);
    bf16_t* Wh0 = (bf16_t*)(ws + off); off += (size_t)H_SZ * H_SZ  * sizeof(bf16_t);
    bf16_t* Wc1 = (bf16_t*)(ws + off); off += (size_t)H_SZ * H_SZ  * sizeof(bf16_t);
    bf16_t* Wh1 = (bf16_t*)(ws + off); off += (size_t)H_SZ * H_SZ  * sizeof(bf16_t);
    float*  b0  = (float*)(ws + off);  off += (size_t)H_SZ * sizeof(float);
    float*  b1  = (float*)(ws + off);  off += (size_t)H_SZ * sizeof(float);
    bf16_t* h0buf = (bf16_t*)(ws + off); off += (size_t)2 * B_SZ * H_SZ  * sizeof(bf16_t);
    bf16_t* h1buf = (bf16_t*)(ws + off); off += (size_t)2 * B_SZ * H_SZ  * sizeof(bf16_t);
    bf16_t* xbuf  = (bf16_t*)(ws + off); off += (size_t)2 * B_SZ * IN_SZ * sizeof(bf16_t);
    int*    bar   = (int*)(ws + off);    off += 2 * sizeof(int);

    const int thr = 256;
    // weight conversions (bf16 copies in natural [N][K] layout)
    cvt_f32_bf16_kernel<<<(H_SZ * IN_SZ + thr - 1) / thr, thr, 0, stream>>>(W_ih0, Wc0, H_SZ * IN_SZ);
    cvt_f32_bf16_kernel<<<(H_SZ * H_SZ + thr - 1) / thr, thr, 0, stream>>>(W_hh0, Wh0, H_SZ * H_SZ);
    cvt_f32_bf16_kernel<<<(H_SZ * H_SZ + thr - 1) / thr, thr, 0, stream>>>(W_ih1, Wc1, H_SZ * H_SZ);
    cvt_f32_bf16_kernel<<<(H_SZ * H_SZ + thr - 1) / thr, thr, 0, stream>>>(W_hh1, Wh1, H_SZ * H_SZ);
    bias_combine_kernel<<<(H_SZ + thr - 1) / thr, thr, 0, stream>>>(b_ih0, b_hh0, b_ih1, b_hh1, b0, b1);
    // zero h0/h1 ping-pong (contiguous bf16: 4*B*H elems = 2*B*H uints) + barrier
    init_state_kernel<<<(2 * B_SZ * H_SZ + thr - 1) / thr, thr, 0, stream>>>(
        (unsigned int*)h0buf, 2 * B_SZ * H_SZ, bar);

    // allow >64KB dynamic LDS (CDNA5 WGP has 320KB), then run the scan
    hipFuncSetAttribute((const void*)rnn_scan_kernel,
                        hipFuncAttributeMaxDynamicSharedMemorySize, LDS_BYTES);
    rnn_scan_kernel<<<NWG, THR, LDS_BYTES, stream>>>(x, Wc0, Wh0, Wc1, Wh1, b0, b1,
                                                     xbuf, h0buf, h1buf, out, hfin, bar);
}
